// RegressionGPModel_2774548873930
// MI455X (gfx1250) — matrix operand order repaired and checked by reading, not compiled, not run on hardware
//
#include <hip/hip_runtime.h>
#include <hip/hip_bf16.h>
#include <math.h>

// Problem constants (from reference setup_inputs)
constexpr int NP = 64;    // P: data_points batch
constexpr int NB = 32;    // B: x batch
constexpr int NA = 40;    // A: atoms per material
constexpr int NAP = 48;   // A padded to 3 full 16-row WMMA tiles
constexpr int ND = 128;   // d: feature dim
constexpr int NPAIR = NP * NB;          // 2048 cross problems
constexpr int NPROB = NPAIR + NB;       // + 32 self problems
constexpr int NTILE = 3;                // 48/16
constexpr int TILES_PER_PROB = NTILE * NTILE;   // 9
constexpr int TOT_TILES = NPROB * TILES_PER_PROB; // 18720

typedef __bf16 bf16_t;
typedef __attribute__((ext_vector_type(16))) __bf16 v16bf;
typedef __attribute__((ext_vector_type(8)))  __bf16 v8bf;
typedef __attribute__((ext_vector_type(8)))  float  v8f;

// ---- workspace layout (bytes, all offsets multiples of 256) ----
constexpr size_t OFF_V1H = 0;                                          // [P*NAP*D] bf16 (rows 40..47 zero)
constexpr size_t OFF_V1L = OFF_V1H + (size_t)NP * NAP * ND * 2;
constexpr size_t OFF_V2H = OFF_V1L + (size_t)NP * NAP * ND * 2;        // [B*NAP*D] bf16
constexpr size_t OFF_V2L = OFF_V2H + (size_t)NB * NAP * ND * 2;
constexpr size_t OFF_S1  = OFF_V2L + (size_t)NB * NAP * ND * 2;        // [P*A] f32
constexpr size_t OFF_S2  = OFF_S1  + (size_t)NP * NA * 4;              // [B*A] f32
constexpr size_t OFF_D2  = OFF_S2  + (size_t)NB * NA * 4;              // [P*B*A*A] f32
constexpr size_t OFF_DS  = OFF_D2  + (size_t)NPAIR * NA * NA * 4;      // [B*A*A] f32
constexpr size_t OFF_KX  = OFF_DS  + (size_t)NB * NA * NA * 4;         // [P*B] f32
constexpr size_t OFF_KXX = OFF_KX  + (size_t)NPAIR * 4;                // [B] f32

// =====================================================================
// Kernel 1: scale inputs by exp(gamma), split to bf16 hi/lo, row norms.
// One block (128 threads) per PADDED row: (P+B)*48 blocks. Pad rows
// (n in [40,48)) are zero-filled so the WMMA kernel needs no guards.
// =====================================================================
__global__ void prep_kernel(const float* __restrict__ x,
                            const float* __restrict__ dp,
                            const float* __restrict__ gamma,
                            bf16_t* __restrict__ v1h, bf16_t* __restrict__ v1l,
                            bf16_t* __restrict__ v2h, bf16_t* __restrict__ v2l,
                            float* __restrict__ s1, float* __restrict__ s2) {
    int row = blockIdx.x;                // 0 .. (NP+NB)*NAP-1
    int t = threadIdx.x;                 // 0..127 == feature index
    int mat = row / NAP;                 // material index (v1 then v2)
    int n   = row - mat * NAP;           // padded atom index

    const float* src;
    bf16_t *oh, *ol;
    float* snorm;
    if (mat < NP) {
        src = dp + ((size_t)mat * NA + n) * ND;
        oh = v1h + (size_t)row * ND; ol = v1l + (size_t)row * ND;
        snorm = s1 + mat * NA + n;
    } else {
        int b = mat - NP;
        src = x + ((size_t)b * NA + n) * ND;
        oh = v2h + ((size_t)b * NAP + n) * ND; ol = v2l + ((size_t)b * NAP + n) * ND;
        snorm = s2 + b * NA + n;
    }

    if (n < NA) {                        // block-uniform branch
        float g = __expf(gamma[t]);
        float val = src[t] * g;
        bf16_t hi = (bf16_t)val;
        bf16_t lo = (bf16_t)(val - (float)hi);
        oh[t] = hi; ol[t] = lo;
        __shared__ float red[128];
        red[t] = val * val;
        __syncthreads();
        for (int s = 64; s > 0; s >>= 1) {
            if (t < s) red[t] += red[t + s];
            __syncthreads();
        }
        if (t == 0) *snorm = red[0];
    } else {                             // zero pad rows
        oh[t] = (bf16_t)0.0f; ol[t] = (bf16_t)0.0f;
    }
}

// =====================================================================
// Kernel 2: pairwise distance tiles via bf16x3-split WMMA.
// One wave per 16x16 output tile; 8 waves per 256-thread block.
// Operands padded to 48 rows -> all fragment loads unconditional,
// EXEC stays all-ones, K-loop is pure load + v_wmma.
// A fragment layout (16-bit 16x32, matched to B):
//   lane half h: elems 0..15 -> K=kk+16h..16h+15
// B fragment layout (16-bit 32x16): lane half h: elems 0..15 -> K=kk+16h..16h+15
// =====================================================================
__device__ inline v16bf load_afrag(const bf16_t* __restrict__ row, int kk, int hf) {
    union { v16bf v; v8bf h[2]; } u;
    u.h[0] = *(const v8bf*)(row + kk + hf * 16);
    u.h[1] = *(const v8bf*)(row + kk + hf * 16 + 8);
    return u.v;
}

__global__ void dist_wmma_kernel(const bf16_t* __restrict__ v1h, const bf16_t* __restrict__ v1l,
                                 const bf16_t* __restrict__ v2h, const bf16_t* __restrict__ v2l,
                                 const float* __restrict__ s1, const float* __restrict__ s2,
                                 float* __restrict__ D2, float* __restrict__ Ds) {
    int wid = (blockIdx.x << 3) + (threadIdx.x >> 5);   // wave id (wave-uniform)
    if (wid >= TOT_TILES) return;                        // uniform; EXEC stays full
    int lane = threadIdx.x & 31;
    int prob = wid / TILES_PER_PROB;
    int tile = wid - prob * TILES_PER_PROB;
    int tn = tile / NTILE, tm = tile - tn * NTILE;

    const bf16_t *ah, *al, *bh, *bl;
    const float *s1p, *s2p;
    float* out;
    if (prob < NPAIR) {
        int p = prob >> 5, b = prob & 31;
        ah = v1h + (size_t)p * NAP * ND; al = v1l + (size_t)p * NAP * ND;
        bh = v2h + (size_t)b * NAP * ND; bl = v2l + (size_t)b * NAP * ND;
        s1p = s1 + p * NA; s2p = s2 + b * NA;
        out = D2 + (size_t)prob * NA * NA;
    } else {                       // self distances: both sides from v2
        int b = prob - NPAIR;
        ah = v2h + (size_t)b * NAP * ND; al = v2l + (size_t)b * NAP * ND;
        bh = ah; bl = al;
        s1p = s2 + b * NA; s2p = s1p;
        out = Ds + (size_t)b * NA * NA;
    }

    int lrow = lane & 15, hf = lane >> 4;
    const bf16_t* arh = ah + (size_t)(tn * 16 + lrow) * ND;  // A row (padded, always valid)
    const bf16_t* arl = al + (size_t)(tn * 16 + lrow) * ND;
    const bf16_t* brh = bh + (size_t)(tm * 16 + lrow) * ND;  // B "column" row
    const bf16_t* brl = bl + (size_t)(tm * 16 + lrow) * ND;

    v8f acc = {};
    #pragma unroll
    for (int kk = 0; kk < ND; kk += 32) {
        v16bf a_hi = load_afrag(arh, kk, hf);
        v16bf a_lo = load_afrag(arl, kk, hf);
        v16bf b_hi = *(const v16bf*)(brh + kk + hf * 16);
        v16bf b_lo = *(const v16bf*)(brl + kk + hf * 16);
        // bf16x3 split product: hi*hi + hi*lo + lo*hi  (~fp32 accuracy)
        acc = __builtin_amdgcn_wmma_f32_16x16x32_bf16(false, a_hi, false, b_hi,
                                                      (short)0, acc, false, false);
        acc = __builtin_amdgcn_wmma_f32_16x16x32_bf16(false, a_hi, false, b_lo,
                                                      (short)0, acc, false, false);
        acc = __builtin_amdgcn_wmma_f32_16x16x32_bf16(false, a_lo, false, b_hi,
                                                      (short)0, acc, false, false);
    }

    // C/D layout: VGPR r, lanes 0-15 -> M=r,N=lane ; lanes 16-31 -> M=r+8,N=lane-16
    int mcol = tm * 16 + lrow;
    float s2v = (mcol < NA) ? s2p[mcol] : 0.f;
    #pragma unroll
    for (int r = 0; r < 8; ++r) {
        int nrow = tn * 16 + r + (hf << 3);
        if (nrow < NA && mcol < NA)
            out[nrow * NA + mcol] = fabsf(s1p[nrow] + s2v - 2.0f * acc[r]);
    }
}

// =====================================================================
// Kernel 3: Hungarian (Jonker-Volgenant shortest augmenting path),
// one 40x40 problem per 64-thread block; column-parallel inner loops.
// Folds the matched exp-sum directly into K_xnew / kxx.
// =====================================================================
__global__ void hungarian_kernel(const float* __restrict__ D2, const float* __restrict__ Ds,
                                 float* __restrict__ Kx, float* __restrict__ kxxv) {
    const float INF = __builtin_inff();
    int prob = blockIdx.x;
    int tid = threadIdx.x;                      // 0..63
    const float* cost = (prob < NPAIR) ? (D2 + (size_t)prob * NA * NA)
                                       : (Ds + (size_t)(prob - NPAIR) * NA * NA);
    __shared__ float sc[NA * NA];
    __shared__ float su[NA + 1], sv[NA + 1], sminv[NA + 1];
    __shared__ int   sp[NA + 1], sway[NA + 1], sused[NA + 1];
    __shared__ float redv[64];
    __shared__ int   redj[64];

    for (int i = tid; i < NA * NA; i += 64) sc[i] = cost[i];
    if (tid <= NA) { su[tid] = 0.f; sv[tid] = 0.f; sp[tid] = 0; sway[tid] = 0; }
    __syncthreads();

    for (int i = 1; i <= NA; ++i) {
        if (tid == 0) sp[0] = i;
        if (tid <= NA) { sminv[tid] = INF; sused[tid] = 0; }
        __syncthreads();
        int j0 = 0;
        for (;;) {
            if (tid == 0) sused[j0] = 1;
            __syncthreads();
            int i0 = sp[j0];
            float u0 = su[i0];
            float best = INF; int bestj = 0;
            if (tid >= 1 && tid <= NA && !sused[tid]) {
                float cur = sc[(i0 - 1) * NA + (tid - 1)] - u0 - sv[tid];
                if (cur < sminv[tid]) { sminv[tid] = cur; sway[tid] = j0; }
                best = sminv[tid]; bestj = tid;
            }
            redv[tid] = best; redj[tid] = bestj;
            __syncthreads();
            for (int s = 32; s > 0; s >>= 1) {
                if (tid < s && redv[tid + s] < redv[tid]) {
                    redv[tid] = redv[tid + s]; redj[tid] = redj[tid + s];
                }
                __syncthreads();
            }
            float delta = redv[0];
            int j1 = redj[0];
            if (tid <= NA) {
                if (sused[tid]) { su[sp[tid]] += delta; sv[tid] -= delta; }
                else            sminv[tid] -= delta;
            }
            __syncthreads();
            j0 = j1;
            if (sp[j0] == 0) break;             // uniform decision
        }
        if (tid == 0) {                          // rewrite augmenting path
            int jj = j0;
            while (jj != 0) { int jn = sway[jj]; sp[jj] = sp[jn]; jj = jn; }
        }
        __syncthreads();
    }

    // matched similarity: (1/A) * sum_j exp(-cost[p[j]-1, j-1])
    float term = 0.f;
    if (tid >= 1 && tid <= NA) term = __expf(-sc[(sp[tid] - 1) * NA + (tid - 1)]);
    redv[tid] = term;
    __syncthreads();
    for (int s = 32; s > 0; s >>= 1) {
        if (tid < s) redv[tid] += redv[tid + s];
        __syncthreads();
    }
    if (tid == 0) {
        float kval = redv[0] / (float)NA;
        if (prob < NPAIR) Kx[prob] = kval;
        else              kxxv[prob - NPAIR] = kval;
    }
}

// =====================================================================
// Kernel 4: finalize. Forward substitution L v = K_xnew (== inv(L)@K),
// mean = K^T alpha, var = kxx - colsum(v*v). One thread per column n.
// =====================================================================
__global__ void finalize_kernel(const float* __restrict__ Kx, const float* __restrict__ kxxv,
                                const float* __restrict__ L, const float* __restrict__ alpha,
                                float* __restrict__ out) {
    int n = threadIdx.x;
    if (n >= NB) return;
    float vcol[NP];
    for (int i = 0; i < NP; ++i) {
        float s = Kx[i * NB + n];
        for (int k = 0; k < i; ++k) s -= L[i * NP + k] * vcol[k];
        vcol[i] = s / L[i * NP + i];
    }
    float mean = 0.f, vv = 0.f;
    for (int i = 0; i < NP; ++i) {
        mean += Kx[i * NB + n] * alpha[i];
        vv   += vcol[i] * vcol[i];
    }
    out[n]      = mean;            // mean[32]
    out[NB + n] = kxxv[n] - vv;    // var[32]
}

// =====================================================================
extern "C" void kernel_launch(void* const* d_in, const int* in_sizes, int n_in,
                              void* d_out, int out_size, void* d_ws, size_t ws_size,
                              hipStream_t stream) {
    const float* x     = (const float*)d_in[0];   // [B,A,D]
    const float* dp    = (const float*)d_in[1];   // [P,A,D]
    const float* gamma = (const float*)d_in[2];   // [1,D]
    const float* L     = (const float*)d_in[3];   // [P,P]
    const float* alpha = (const float*)d_in[4];   // [P]
    float* out = (float*)d_out;                   // [mean(32) | var(32)]

    char* ws = (char*)d_ws;
    bf16_t* v1h = (bf16_t*)(ws + OFF_V1H);
    bf16_t* v1l = (bf16_t*)(ws + OFF_V1L);
    bf16_t* v2h = (bf16_t*)(ws + OFF_V2H);
    bf16_t* v2l = (bf16_t*)(ws + OFF_V2L);
    float*  s1  = (float*)(ws + OFF_S1);
    float*  s2  = (float*)(ws + OFF_S2);
    float*  D2  = (float*)(ws + OFF_D2);
    float*  Ds  = (float*)(ws + OFF_DS);
    float*  Kx  = (float*)(ws + OFF_KX);
    float*  kxx = (float*)(ws + OFF_KXX);

    // 1) scale + bf16 split + norms (writes padded operand buffers fully)
    prep_kernel<<<(NP + NB) * NAP, ND, 0, stream>>>(x, dp, gamma,
                                                    v1h, v1l, v2h, v2l, s1, s2);
    // 2) WMMA distance tiles: 18720 wave-tiles, 8 waves per block
    dist_wmma_kernel<<<TOT_TILES / 8, 256, 0, stream>>>(v1h, v1l, v2h, v2l,
                                                        s1, s2, D2, Ds);
    // 3) Hungarian per problem
    hungarian_kernel<<<NPROB, 64, 0, stream>>>(D2, Ds, Kx, kxx);
    // 4) triangular solve + mean/var
    finalize_kernel<<<1, 32, 0, stream>>>(Kx, kxx, L, alpha, out);
}
